// Whole_NEUMF_Sample_40699110097057
// MI455X (gfx1250) — matrix operand order repaired and checked by reading, not compiled, not run on hardware
//
#include <hip/hip_runtime.h>
#include <hip/hip_bf16.h>

#define NN 8192
#define DD 128

typedef __attribute__((ext_vector_type(16))) __bf16 v16bf;
typedef __attribute__((ext_vector_type(8)))  __bf16 v8bf;
typedef __attribute__((ext_vector_type(8)))  float  v8f;

// ---------------------------------------------------------------------------
// Kernel 1: va[k] = sum_j W2a[k][j]*w3a[j]  (t<128) ; vb[k-128] for t>=128
// ---------------------------------------------------------------------------
__global__ void prep_vec_kernel(const float* __restrict__ w2,
                                const float* __restrict__ w3,
                                float* __restrict__ va_vb) {
    int t = blockIdx.x * blockDim.x + threadIdx.x;   // 0..255
    if (t >= 2 * DD) return;
    const float* row = w2 + (size_t)t * DD;
    float s = 0.0f;
#pragma unroll 8
    for (int j = 0; j < DD; ++j) s = fmaf(row[j], w3[j], s);
    va_vb[t] = s;
}

// ---------------------------------------------------------------------------
// Kernel 2: per-row bias terms a[i], b[i] (f32) + bf16 copies of z and z*w3b
// ---------------------------------------------------------------------------
__global__ void prep_rows_kernel(const float* __restrict__ z,
                                 const float* __restrict__ w3,
                                 const float* __restrict__ va_vb,
                                 __bf16* __restrict__ zb,
                                 __bf16* __restrict__ zwb,
                                 float* __restrict__ arow,
                                 float* __restrict__ brow) {
    const int i = blockIdx.x;      // row
    const int k = threadIdx.x;     // 0..127
    const size_t idx = (size_t)i * DD + k;
    float zk = z[idx];
    zb[idx]  = (__bf16)zk;
    zwb[idx] = (__bf16)(zk * w3[DD + k]);   // w3b = weight_three[128:]
    float rz = fmaxf(zk, 0.0f);

    __shared__ float sa[DD];
    __shared__ float sb[DD];
    sa[k] = rz * va_vb[k];
    sb[k] = rz * va_vb[DD + k];
    __syncthreads();
    for (int off = DD / 2; off > 0; off >>= 1) {
        if (k < off) { sa[k] += sa[k + off]; sb[k] += sb[k + off]; }
        __syncthreads();
    }
    if (k == 0) { arow[i] = sa[0]; brow[i] = sb[0]; }
}

// ---------------------------------------------------------------------------
// Kernel 3: out[i][j] = sigmoid(a[i] + b[j] + sum_k zw[i][k]*z[j][k])
// 8 waves/block; each wave computes a 32x32 tile via 2x2 v_wmma_f32_16x16x32_bf16.
// Block tile = 128 rows x 64 cols.
// ---------------------------------------------------------------------------
__global__ __launch_bounds__(256) void neumf_main_kernel(
    const __bf16* __restrict__ zwb,   // A source: (z * w3b), row-major N x D
    const __bf16* __restrict__ zb,    // B source: z, row-major N x D (cols of z^T)
    const float*  __restrict__ arow,
    const float*  __restrict__ brow,
    float* __restrict__ out) {
    const int lane    = threadIdx.x & 31;
    const int wave    = threadIdx.x >> 5;     // 0..7
    const int waveRow = wave & 3;             // 4 row-tiles of 32
    const int waveCol = wave >> 2;            // 2 col-tiles of 32
    const int rowBase = blockIdx.y * 128 + waveRow * 32;
    const int colBase = blockIdx.x * 64  + waveCol * 32;
    const int m16  = lane & 15;
    const int half = lane >> 4;               // 0: lanes 0-15, 1: lanes 16-31

    // A fragment rows (zw) and B fragment rows (z; they are columns of z^T)
    const __bf16* aPtr0 = zwb + (size_t)(rowBase + m16) * DD;
    const __bf16* aPtr1 = aPtr0 + (size_t)16 * DD;
    const __bf16* bPtr0 = zb  + (size_t)(colBase + m16) * DD;
    const __bf16* bPtr1 = bPtr0 + (size_t)16 * DD;

    v8f acc[2][2] = {};

#pragma unroll
    for (int kk = 0; kk < DD; kk += 32) {
        // A 16x32, contiguous k order used here (B matches): lanes 0-15 hold K 0-7 (lo) + 8-15 (hi),
        // lanes 16-31 hold K 16-23 (lo) + 24-31 (hi).
        v8bf a0lo = *(const v8bf*)(aPtr0 + kk + half * 16);
        v8bf a0hi = *(const v8bf*)(aPtr0 + kk + 8 + half * 16);
        v8bf a1lo = *(const v8bf*)(aPtr1 + kk + half * 16);
        v8bf a1hi = *(const v8bf*)(aPtr1 + kk + 8 + half * 16);
        v16bf af0 = __builtin_shufflevector(a0lo, a0hi,
                      0,1,2,3,4,5,6,7,8,9,10,11,12,13,14,15);
        v16bf af1 = __builtin_shufflevector(a1lo, a1hi,
                      0,1,2,3,4,5,6,7,8,9,10,11,12,13,14,15);
        // B 32x16, same contiguous k order as A: lane holds column (lane%16) with 16 consecutive K
        // values (lanes 0-15: K 0-15, lanes 16-31: K 16-31) -> one 32B load.
        v16bf bf0 = *(const v16bf*)(bPtr0 + kk + half * 16);
        v16bf bf1 = *(const v16bf*)(bPtr1 + kk + half * 16);

        acc[0][0] = __builtin_amdgcn_wmma_f32_16x16x32_bf16(
            false, af0, false, bf0, (short)0, acc[0][0], false, false);
        acc[0][1] = __builtin_amdgcn_wmma_f32_16x16x32_bf16(
            false, af0, false, bf1, (short)0, acc[0][1], false, false);
        acc[1][0] = __builtin_amdgcn_wmma_f32_16x16x32_bf16(
            false, af1, false, bf0, (short)0, acc[1][0], false, false);
        acc[1][1] = __builtin_amdgcn_wmma_f32_16x16x32_bf16(
            false, af1, false, bf1, (short)0, acc[1][1], false, false);
    }

    // C/D layout: VGPR v, lanes 0-15 -> M=v, N=lane; lanes 16-31 -> M=v+8.
#pragma unroll
    for (int c = 0; c < 2; ++c) {
        const int col = colBase + c * 16 + m16;
        const float bc = brow[col];
#pragma unroll
        for (int r = 0; r < 2; ++r) {
            const int rtop = rowBase + r * 16 + half * 8;
#pragma unroll
            for (int v = 0; v < 8; ++v) {
                const int row = rtop + v;
                float s = acc[r][c][v] + arow[row] + bc;
                out[(size_t)row * NN + col] = 1.0f / (1.0f + __expf(-s));
            }
        }
    }
}

// ---------------------------------------------------------------------------
// Host launcher
// ---------------------------------------------------------------------------
extern "C" void kernel_launch(void* const* d_in, const int* in_sizes, int n_in,
                              void* d_out, int out_size, void* d_ws, size_t ws_size,
                              hipStream_t stream) {
    const float* z  = (const float*)d_in[0];   // (8192, 128)
    const float* w2 = (const float*)d_in[1];   // (256, 128)
    const float* w3 = (const float*)d_in[2];   // (256, 1)
    float* out = (float*)d_out;                // (8192, 8192)

    char* ws = (char*)d_ws;
    __bf16* zb    = (__bf16*)(ws);                                   // 2 MB
    __bf16* zwb   = (__bf16*)(ws + (size_t)NN * DD * 2);             // 2 MB
    float*  va_vb = (float*) (ws + (size_t)NN * DD * 4);             // 1 KB
    float*  arow  = (float*) (ws + (size_t)NN * DD * 4 + 4096);      // 32 KB
    float*  brow  = arow + NN;                                       // 32 KB

    prep_vec_kernel<<<1, 256, 0, stream>>>(w2, w3, va_vb);
    prep_rows_kernel<<<NN, DD, 0, stream>>>(z, w3, va_vb, zb, zwb, arow, brow);
    neumf_main_kernel<<<dim3(NN / 64, NN / 128), 256, 0, stream>>>(
        zwb, zb, arow, brow, out);
}
